// IDDecoder_29815662969443
// MI455X (gfx1250) — compile-verified, matrix operand order checked by reading, not run on hardware
//
#include <hip/hip_runtime.h>
#include <hip/hip_bf16.h>

// ---------------------------------------------------------------------------
// Shapes (compile-time constants from the reference)
// ---------------------------------------------------------------------------
#define BGc   32        // B*G
#define TLc   1800      // T*N trajectory tokens per bg
#define TLPc  1824      // padded key length (multiple of 32; col 1800 = zero-attn)
#define QLc   60        // CT*CN query tokens per bg
#define QLPc  64        // padded query rows (multiple of 16)
#define Ec    512       // embed dim
#define Hc    8         // heads
#define Dc    64        // head dim
#define Lc    6         // layers
#define VP1c  51
#define PEc   32
#define DFFc  2048
#define MTOT  (BGc*QLc) // 1920 query tokens total
#define MTRAJ (BGc*TLc) // 57600 traj tokens total

typedef __attribute__((ext_vector_type(16))) __bf16 v16bf;
typedef __attribute__((ext_vector_type(8)))  float  v8f;
typedef int v4i __attribute__((vector_size(16)));

#if __has_builtin(__builtin_amdgcn_global_load_async_to_lds_b128)
#define HAVE_ASYNC 1
#else
#define HAVE_ASYNC 0
#endif

__device__ __forceinline__ unsigned short f2bf(float f) {
  unsigned int u = __float_as_uint(f);
  u = (u + 0x7FFFu + ((u >> 16) & 1u)) >> 16;
  return (unsigned short)u;
}

__device__ __forceinline__ v16bf ldbf16(const unsigned short* p) {
  union { v16bf v; uint4 q[2]; } u;
  const uint4* s = (const uint4*)p;
  u.q[0] = s[0]; u.q[1] = s[1];
  return u.v;
}
__device__ __forceinline__ v16bf zbf16() {
  union { v16bf v; uint4 q[2]; } u;
  u.q[0] = make_uint4(0u,0u,0u,0u); u.q[1] = make_uint4(0u,0u,0u,0u);
  return u.v;
}

#define WMMA_BF16(a,b,c) \
  __builtin_amdgcn_wmma_f32_16x16x32_bf16(false,(a),false,(b),(short)0,(c),false,false)

// 16-byte Global -> LDS copy, async (ASYNCcnt-tracked path) if available.
__device__ __forceinline__ void async_cp16(const unsigned short* g, unsigned short* l) {
#if HAVE_ASYNC
  __builtin_amdgcn_global_load_async_to_lds_b128(
      (v4i*)(void*)const_cast<unsigned short*>(g), (v4i*)(void*)l, 0, 0);
#else
  *(uint4*)l = *(const uint4*)g;
#endif
}

__device__ __forceinline__ void wait_async0() {
#if HAVE_ASYNC
#if __has_builtin(__builtin_amdgcn_s_wait_asynccnt)
  __builtin_amdgcn_s_wait_asynccnt(0);
#else
  asm volatile("s_wait_asynccnt 0x0" ::: "memory");
#endif
#endif
}

// ---------------------------------------------------------------------------
// Shared GEMM epilogue: bias, optional exact GELU, f32 and/or bf16 stores.
// Epilogue modes: 0 = normal, 1 = remap rows m -> (m/60)*64 + m%60 (padded Q),
//                 2 = transposed bf16 out (V^T: out[(m/1800)*512+n][m%1800])
// ---------------------------------------------------------------------------
__device__ __forceinline__ void gemm_epilogue(
    v8f c00, v8f c01, v8f c10, v8f c11, int m0, int n0, int lane,
    const float* __restrict__ bias, float* __restrict__ outF, int ldoF,
    unsigned short* __restrict__ outB, int ldoB, int mode, int doGelu)
{
  const int rowAdd = (lane >= 16) ? 8 : 0;
  const int col = lane & 15;
#pragma unroll
  for (int mi = 0; mi < 2; mi++) {
#pragma unroll
    for (int ni = 0; ni < 2; ni++) {
      v8f acc = (mi == 0) ? ((ni == 0) ? c00 : c01) : ((ni == 0) ? c10 : c11);
      const int ncol = n0 + ni * 16 + col;
      const float bv = bias ? bias[ncol] : 0.0f;
#pragma unroll
      for (int i = 0; i < 8; i++) {
        const int row = m0 + mi * 16 + i + rowAdd;
        float v = acc[i] + bv;
        if (doGelu) v = 0.5f * v * (1.0f + erff(v * 0.70710678118f));
        if (outF) outF[(long long)row * ldoF + ncol] = v;
        if (outB) {
          long long idx;
          if (mode == 1)      idx = (long long)((row / 60) * 64 + (row % 60)) * ldoB + ncol;
          else if (mode == 2) idx = ((long long)(row / TLc) * Ec + ncol) * ldoB + (row % TLc);
          else                idx = (long long)row * ldoB + ncol;
          outB[idx] = f2bf(v);
        }
      }
    }
  }
}

// ---------------------------------------------------------------------------
// LDS-staged bf16 WMMA GEMM:  C[M,N] = A[M,K] * W[N,K]^T + bias
// Workgroup = 256 threads (8 waves, 4x2 wave grid) computes a 128x64 tile.
// Per 32-wide k-step, a 128x32 A tile + 64x32 B tile are staged into LDS via
// async Global->LDS b128 copies (double-buffered, s_wait_asynccnt + barrier),
// then each wave runs 2x2 v_wmma_f32_16x16x32_bf16 from ds_load_b128 reads.
// LDS rows padded to 40 elems (bank stride 20 -> conflict-free 16-lane reads).
// ---------------------------------------------------------------------------
#define TMg 128
#define TNg 64
#define LDS_STRIDE 40
#define A_BUF_ELEMS (TMg * LDS_STRIDE)
#define B_BUF_ELEMS (TNg * LDS_STRIDE)
#define B_BASE      (2 * A_BUF_ELEMS)

__global__ __launch_bounds__(256)
void gemm_bf16_kernel(const unsigned short* __restrict__ A, int lda,
                      const unsigned short* __restrict__ W, int ldw,
                      const float* __restrict__ bias, int K,
                      float* __restrict__ outF, int ldoF,
                      unsigned short* __restrict__ outB, int ldoB,
                      int mode, int doGelu)
{
  __shared__ unsigned short smem[2 * A_BUF_ELEMS + 2 * B_BUF_ELEMS];
  const int M0 = blockIdx.x * TMg, N0 = blockIdx.y * TNg;
  const int tid = threadIdx.x;
  const int wave = tid >> 5, lane = tid & 31;
  const int mw = wave & 3, nw = wave >> 2;    // 4 x 2 waves over 128x64
  const int half = lane >> 4, r = lane & 15;

  // stage one 32-wide k-slab: A = 512 b128 chunks (2/thread), B = 256 (1/thread)
  auto stage = [&](int k0, int buf) {
    unsigned short* Ad = smem + buf * A_BUF_ELEMS;
    unsigned short* Bd = smem + B_BASE + buf * B_BUF_ELEMS;
    int c = tid, row = c >> 2, q = c & 3;
    async_cp16(A + (long long)(M0 + row) * lda + k0 + q * 8,
               Ad + row * LDS_STRIDE + q * 8);
    c = tid + 256; row = c >> 2; q = c & 3;
    async_cp16(A + (long long)(M0 + row) * lda + k0 + q * 8,
               Ad + row * LDS_STRIDE + q * 8);
    c = tid; row = c >> 2; q = c & 3;
    async_cp16(W + (long long)(N0 + row) * ldw + k0 + q * 8,
               Bd + row * LDS_STRIDE + q * 8);
  };

  v8f c00 = {}, c01 = {}, c10 = {}, c11 = {};
  stage(0, 0);
  const int S = K >> 5;
  for (int s = 0; s < S; s++) {
    wait_async0();        // this wave's staged slab s is in LDS
    __syncthreads();      // ...and everyone else's; prior-buffer reads done
    if (s + 1 < S) stage((s + 1) << 5, (s + 1) & 1);
    const unsigned short* Ap = smem + (s & 1) * A_BUF_ELEMS;
    const unsigned short* Bp = smem + B_BASE + (s & 1) * B_BUF_ELEMS;
    v16bf a0 = ldbf16(Ap + (mw * 32 + r)      * LDS_STRIDE + half * 16);
    v16bf a1 = ldbf16(Ap + (mw * 32 + 16 + r) * LDS_STRIDE + half * 16);
    v16bf b0 = ldbf16(Bp + (nw * 32 + r)      * LDS_STRIDE + half * 16);
    v16bf b1 = ldbf16(Bp + (nw * 32 + 16 + r) * LDS_STRIDE + half * 16);
    c00 = WMMA_BF16(a0, b0, c00);
    c01 = WMMA_BF16(a0, b1, c01);
    c10 = WMMA_BF16(a1, b0, c10);
    c11 = WMMA_BF16(a1, b1, c11);
  }
  gemm_epilogue(c00, c01, c10, c11, M0 + mw * 32, N0 + nw * 32, lane,
                bias, outF, ldoF, outB, ldoB, mode, doGelu);
}

// ---------------------------------------------------------------------------
// Cross-attn score GEMM (WMMA, K = head dim 64) with fused bias:
//   causal (tt>=ut -> -inf), key padding mask, rel-PE gather, zero-attn column.
// ---------------------------------------------------------------------------
__global__ __launch_bounds__(32)
void attn_score_kernel(const unsigned short* __restrict__ qbf,   // [BG,64,E]
                       const unsigned short* __restrict__ kbf,   // [BG,1800,E]
                       const int* __restrict__ tt, const int* __restrict__ ut,
                       const unsigned char* __restrict__ tm,
                       const float* __restrict__ rpe,            // [32,8] layer slice
                       float* __restrict__ scores)               // [BG*H,64,1824]
{
  const int kt = blockIdx.x, qt = blockIdx.y, bh = blockIdx.z;
  const int bg = bh >> 3, h = bh & 7;
  const int lane = threadIdx.x, half = lane >> 4, r = lane & 15;
  v8f acc = {};
  const unsigned short* qb = qbf + ((long long)bg * QLPc + qt * 16 + r) * Ec + h * Dc + half * 16;
  const int tok_b = kt * 16 + r;
  const int tok_c = (tok_b < TLc) ? tok_b : (TLc - 1);
  const unsigned short* kb = kbf + ((long long)bg * TLc + tok_c) * Ec + h * Dc + half * 16;
#pragma unroll
  for (int kk = 0; kk < Dc; kk += 32) {
    v16bf a = ldbf16(qb + kk);
    v16bf b = (tok_b < TLc) ? ldbf16(kb + kk) : zbf16();
    acc = WMMA_BF16(a, b, acc);
  }
  const int rowAdd = (lane >= 16) ? 8 : 0;
  const int col = lane & 15;
  float* srow = scores + (long long)bh * QLPc * TLPc;
#pragma unroll
  for (int i = 0; i < 8; i++) {
    const int q = qt * 16 + i + rowAdd;
    const int tok = kt * 16 + col;
    float s;
    if (tok < TLc) {
      const int ttv = tt[bg * TLc + tok];
      const int qc = (q < QLc) ? q : (QLc - 1);
      const int utv = ut[bg * QLc + qc];
      if (ttv >= utv || tm[bg * TLc + tok]) s = -INFINITY;
      else s = acc[i] * 0.125f + rpe[((utv - ttv) & (PEc - 1)) * Hc + h];
    } else if (tok == TLc) s = 0.0f;   // add_zero_attn column: q.0/sqrt(d)+0
    else s = -INFINITY;                 // pad to 1824
    srow[(long long)q * TLPc + tok] = s;
  }
}

// Row softmax over 1824 logits -> bf16 probabilities.
__global__ __launch_bounds__(256)
void softmax_kernel(const float* __restrict__ scores, unsigned short* __restrict__ p)
{
  const int row = blockIdx.x;                 // 0 .. BG*H*60-1
  const long long base = ((long long)(row / QLc) * QLPc + (row % QLc)) * TLPc;
  const float* s = scores + base;
  unsigned short* pr = p + base;
  __shared__ float red[256];
  const int tid = threadIdx.x;
  float mx = -INFINITY;
  for (int c = tid; c < TLPc; c += 256) mx = fmaxf(mx, s[c]);
  red[tid] = mx; __syncthreads();
  for (int st = 128; st > 0; st >>= 1) { if (tid < st) red[tid] = fmaxf(red[tid], red[tid + st]); __syncthreads(); }
  mx = red[0]; __syncthreads();
  float sum = 0.f;
  for (int c = tid; c < TLPc; c += 256) sum += expf(s[c] - mx);
  red[tid] = sum; __syncthreads();
  for (int st = 128; st > 0; st >>= 1) { if (tid < st) red[tid] += red[tid + st]; __syncthreads(); }
  const float inv = 1.0f / red[0];
  for (int c = tid; c < TLPc; c += 256) pr[c] = f2bf(expf(s[c] - mx) * inv);
}

// P @ V  (WMMA, K = 1824) against transposed V.
__global__ __launch_bounds__(32)
void attn_pv_kernel(const unsigned short* __restrict__ p,   // [BG*H,64,1824]
                    const unsigned short* __restrict__ vt,  // [BG,E,1824]
                    unsigned short* __restrict__ obf)       // [1920,512] bf16
{
  const int dt = blockIdx.x, qt = blockIdx.y, bh = blockIdx.z;
  const int bg = bh >> 3, h = bh & 7;
  const int lane = threadIdx.x, half = lane >> 4, r = lane & 15;
  v8f acc = {};
  const unsigned short* pb = p  + ((long long)bh * QLPc + qt * 16 + r) * TLPc + half * 16;
  const unsigned short* vb = vt + ((long long)(bg * Ec + h * Dc + dt * 16 + r)) * TLPc + half * 16;
  for (int kk = 0; kk < TLPc; kk += 32) {
    v16bf a = ldbf16(pb + kk);
    v16bf b = ldbf16(vb + kk);
    __builtin_prefetch(pb + kk + 64, 0, 3);
    acc = WMMA_BF16(a, b, acc);
  }
  const int rowAdd = (lane >= 16) ? 8 : 0;
  const int col = lane & 15;
#pragma unroll
  for (int i = 0; i < 8; i++) {
    const int q = qt * 16 + i + rowAdd;
    if (q < QLc) {
      const int d = h * Dc + dt * 16 + col;
      obf[((long long)(bg * QLc + q)) * Ec + d] = f2bf(acc[i]);
    }
  }
}

// ---------------------------------------------------------------------------
// Small VALU kernels
// ---------------------------------------------------------------------------
__global__ __launch_bounds__(256)
void f32_to_bf16_kernel(const float* __restrict__ in, unsigned short* __restrict__ out, int n)
{
  const int i = blockIdx.x * 256 + threadIdx.x;
  if (i < n) out[i] = f2bf(in[i]);
}

__global__ __launch_bounds__(256)
void build_traj_kernel(const float* __restrict__ tf, const int* __restrict__ ids,
                       const float* __restrict__ Ww, unsigned short* __restrict__ traj)
{
  const long long tok = blockIdx.x; const int tid = threadIdx.x;
  traj[tok * Ec + tid]       = f2bf(tf[tok * 256 + tid]);
  const int id = ids[tok];
  traj[tok * Ec + 256 + tid] = f2bf(Ww[tid * VP1c + id]);   // W_word.T[id]
}

__global__ __launch_bounds__(256)
void build_unk_kernel(const float* __restrict__ uf, const float* __restrict__ Ww,
                      float* __restrict__ xf, unsigned short* __restrict__ xbf)
{
  const long long tok = blockIdx.x; const int tid = threadIdx.x;
  const float a = uf[tok * 256 + tid];
  const float b = Ww[tid * VP1c + (VP1c - 1)];              // unk embedding
  xf[tok * Ec + tid] = a;        xf[tok * Ec + 256 + tid] = b;
  xbf[tok * Ec + tid] = f2bf(a); xbf[tok * Ec + 256 + tid] = f2bf(b);
}

// Tiny self-attention core over 60 keys (+ zero-attn column), per (q,h,bg).
__global__ __launch_bounds__(64)
void sa_core_kernel(const float* __restrict__ qf, const float* __restrict__ kf,
                    const float* __restrict__ vf, const unsigned char* __restrict__ um,
                    unsigned short* __restrict__ obf)
{
  const int q = blockIdx.x, h = blockIdx.y, bg = blockIdx.z;
  const int tid = threadIdx.x;
  __shared__ float qv[64]; __shared__ float pl[64]; __shared__ float red[64];
  const long long tokq = (long long)(bg * QLc + q);
  qv[tid] = qf[tokq * Ec + h * Dc + tid];
  __syncthreads();
  float s = -INFINITY;
  if (tid < QLc) {
    const float* kr = kf + ((long long)(bg * QLc + tid)) * Ec + h * Dc;
    float d = 0.f;
#pragma unroll
    for (int j = 0; j < Dc; j++) d += qv[j] * kr[j];
    s = d * 0.125f;
    if (um[bg * QLc + tid]) s = -INFINITY;
  }
  red[tid] = s; __syncthreads();
  for (int st = 32; st > 0; st >>= 1) { if (tid < st) red[tid] = fmaxf(red[tid], red[tid + st]); __syncthreads(); }
  const float mx = fmaxf(red[0], 0.0f);   // include zero-attn logit 0
  __syncthreads();
  const float e = (tid < QLc) ? expf(s - mx) : 0.f;
  red[tid] = e; __syncthreads();
  for (int st = 32; st > 0; st >>= 1) { if (tid < st) red[tid] += red[tid + st]; __syncthreads(); }
  const float den = red[0] + expf(-mx);   // zero-attn column in denominator
  __syncthreads();
  pl[tid] = e / den;
  __syncthreads();
  float o = 0.f;
  for (int j = 0; j < QLc; j++) o += pl[j] * vf[((long long)(bg * QLc + j)) * Ec + h * Dc + tid];
  obf[tokq * Ec + h * Dc + tid] = f2bf(o);
}

// x = LN(x + y) * g + b ; also emit bf16 copy for next WMMA GEMM.
__global__ __launch_bounds__(256)
void resid_ln_kernel(float* __restrict__ xf, const float* __restrict__ yf,
                     const float* __restrict__ g, const float* __restrict__ b,
                     unsigned short* __restrict__ xbf)
{
  const int t = blockIdx.x, tid = threadIdx.x;
  const long long base = (long long)t * Ec;
  const float v0 = xf[base + tid]       + yf[base + tid];
  const float v1 = xf[base + 256 + tid] + yf[base + 256 + tid];
  __shared__ float red[256];
  red[tid] = v0 + v1; __syncthreads();
  for (int st = 128; st > 0; st >>= 1) { if (tid < st) red[tid] += red[tid + st]; __syncthreads(); }
  const float mean = red[0] * (1.0f / (float)Ec); __syncthreads();
  const float d0 = v0 - mean, d1 = v1 - mean;
  red[tid] = d0 * d0 + d1 * d1; __syncthreads();
  for (int st = 128; st > 0; st >>= 1) { if (tid < st) red[tid] += red[tid + st]; __syncthreads(); }
  const float inv = rsqrtf(red[0] * (1.0f / (float)Ec) + 1e-5f);
  const float o0 = d0 * inv * g[tid] + b[tid];
  const float o1 = d1 * inv * g[256 + tid] + b[256 + tid];
  xf[base + tid] = o0;        xf[base + 256 + tid] = o1;
  xbf[base + tid] = f2bf(o0); xbf[base + 256 + tid] = f2bf(o1);
}

// logits = x[:, 256:] @ W_out^T  (N=51, tiny)
__global__ __launch_bounds__(64)
void head_kernel(const float* __restrict__ xf, const float* __restrict__ Wout,
                 float* __restrict__ out)
{
  const int t = blockIdx.x, v = threadIdx.x;
  if (v >= VP1c) return;
  const float* x = xf + (long long)t * Ec + 256;
  const float* wr = Wout + v * 256;
  float s = 0.f;
  for (int d = 0; d < 256; d++) s += x[d] * wr[d];
  out[t * VP1c + v] = s;
}

// ---------------------------------------------------------------------------
// Host launcher
// ---------------------------------------------------------------------------
static void launch_gemm(hipStream_t st, const unsigned short* A, int lda,
                        const unsigned short* W, int ldw, const float* bias,
                        int M, int N, int K, float* outF, int ldoF,
                        unsigned short* outB, int ldoB, int mode, int gelu)
{
  dim3 grid(M / TMg, N / TNg);
  gemm_bf16_kernel<<<grid, 256, 0, st>>>(A, lda, W, ldw, bias, K, outF, ldoF, outB, ldoB, mode, gelu);
}

extern "C" void kernel_launch(void* const* d_in, const int* in_sizes, int n_in,
                              void* d_out, int out_size, void* d_ws, size_t ws_size,
                              hipStream_t stream)
{
  (void)in_sizes; (void)n_in; (void)out_size; (void)ws_size;
  // ---- inputs (setup_inputs order) ----
  const float* tf   = (const float*)d_in[0];
  const float* uf   = (const float*)d_in[1];
  const int*   ids  = (const int*)d_in[2];
  const int*   tt   = (const int*)d_in[3];
  const int*   ut   = (const int*)d_in[4];
  const unsigned char* tm = (const unsigned char*)d_in[5];
  const unsigned char* um = (const unsigned char*)d_in[6];
  const float* Ww   = (const float*)d_in[8];
  const float* Wout = (const float*)d_in[9];
  const float* rpe  = (const float*)d_in[10];
  const float* saWqkv = (const float*)d_in[11];
  const float* sabqkv = (const float*)d_in[12];
  const float* saWo   = (const float*)d_in[13];
  const float* sabo   = (const float*)d_in[14];
  const float* salng  = (const float*)d_in[15];
  const float* salnb  = (const float*)d_in[16];
  const float* caWqkv = (const float*)d_in[17];
  const float* cabqkv = (const float*)d_in[18];
  const float* caWo   = (const float*)d_in[19];
  const float* cabo   = (const float*)d_in[20];
  const float* calng  = (const float*)d_in[21];
  const float* calnb  = (const float*)d_in[22];
  const float* f1     = (const float*)d_in[23];
  const float* fb1    = (const float*)d_in[24];
  const float* f2     = (const float*)d_in[25];
  const float* fb2    = (const float*)d_in[26];
  const float* flng   = (const float*)d_in[27];
  const float* flnb   = (const float*)d_in[28];
  float* out = (float*)d_out;

  // ---- workspace carve ----
  unsigned char* wsb = (unsigned char*)d_ws;
  size_t off = 0;
  auto carve = [&](size_t bytes) -> void* {
    void* p = wsb + off; off += (bytes + 255) & ~(size_t)255; return p;
  };
  unsigned short* traj_bf = (unsigned short*)carve((size_t)MTRAJ * Ec * 2);
  unsigned short* k_bf    = (unsigned short*)carve((size_t)MTRAJ * Ec * 2);
  unsigned short* vt_bf   = (unsigned short*)carve((size_t)BGc * Ec * TLPc * 2);
  float*          scores  = (float*)carve((size_t)BGc * Hc * QLPc * TLPc * 4);
  unsigned short* p_bf    = (unsigned short*)carve((size_t)BGc * Hc * QLPc * TLPc * 2);
  float*          xf      = (float*)carve((size_t)MTOT * Ec * 4);
  unsigned short* xbf     = (unsigned short*)carve((size_t)MTOT * Ec * 2);
  float*          yf      = (float*)carve((size_t)MTOT * Ec * 4);
  unsigned short* qpad    = (unsigned short*)carve((size_t)BGc * QLPc * Ec * 2);
  unsigned short* obf     = (unsigned short*)carve((size_t)MTOT * Ec * 2);
  float*          saqf    = (float*)carve((size_t)MTOT * Ec * 4);
  float*          sakf    = (float*)carve((size_t)MTOT * Ec * 4);
  float*          savf    = (float*)carve((size_t)MTOT * Ec * 4);
  unsigned short* h1bf    = (unsigned short*)carve((size_t)MTOT * DFFc * 2);
  unsigned short* wsaqkv  = (unsigned short*)carve((size_t)(Lc-1) * 3 * Ec * Ec * 2);
  unsigned short* wsao    = (unsigned short*)carve((size_t)(Lc-1) * Ec * Ec * 2);
  unsigned short* wcaqkv  = (unsigned short*)carve((size_t)Lc * 3 * Ec * Ec * 2);
  unsigned short* wcao    = (unsigned short*)carve((size_t)Lc * Ec * Ec * 2);
  unsigned short* wf1     = (unsigned short*)carve((size_t)Lc * DFFc * Ec * 2);
  unsigned short* wf2     = (unsigned short*)carve((size_t)Lc * Ec * DFFc * 2);

  // ---- weight f32 -> bf16 ----
  auto conv = [&](const float* src, unsigned short* dst, int n) {
    f32_to_bf16_kernel<<<(n + 255) / 256, 256, 0, stream>>>(src, dst, n);
  };
  conv(saWqkv, wsaqkv, (Lc-1) * 3 * Ec * Ec);
  conv(saWo,   wsao,   (Lc-1) * Ec * Ec);
  conv(caWqkv, wcaqkv, Lc * 3 * Ec * Ec);
  conv(caWo,   wcao,   Lc * Ec * Ec);
  conv(f1,     wf1,    Lc * DFFc * Ec);
  conv(f2,     wf2,    Lc * Ec * DFFc);

  // ---- zero the padded regions once (pad rows/cols never rewritten) ----
  (void)hipMemsetAsync(qpad, 0, (size_t)BGc * QLPc * Ec * 2, stream);
  (void)hipMemsetAsync(vt_bf, 0, (size_t)BGc * Ec * TLPc * 2, stream);
  (void)hipMemsetAsync(p_bf, 0, (size_t)BGc * Hc * QLPc * TLPc * 2, stream);

  // ---- embed + concat ----
  build_traj_kernel<<<MTRAJ, 256, 0, stream>>>(tf, ids, Ww, traj_bf);
  build_unk_kernel<<<MTOT, 256, 0, stream>>>(uf, Ww, xf, xbf);

  for (int l = 0; l < Lc; l++) {
    if (l > 0) {
      const int j = l - 1;
      const unsigned short* Wq = wsaqkv + (size_t)j * 3 * Ec * Ec;
      // self-attention QKV projections (WMMA)
      launch_gemm(stream, xbf, Ec, Wq,                 Ec, sabqkv + j*3*Ec,          MTOT, Ec, Ec, saqf, Ec, nullptr, 0, 0, 0);
      launch_gemm(stream, xbf, Ec, Wq + (size_t)Ec*Ec, Ec, sabqkv + j*3*Ec + Ec,     MTOT, Ec, Ec, sakf, Ec, nullptr, 0, 0, 0);
      launch_gemm(stream, xbf, Ec, Wq + (size_t)2*Ec*Ec, Ec, sabqkv + j*3*Ec + 2*Ec, MTOT, Ec, Ec, savf, Ec, nullptr, 0, 0, 0);
      sa_core_kernel<<<dim3(QLc, Hc, BGc), 64, 0, stream>>>(saqf, sakf, savf, um, obf);
      launch_gemm(stream, obf, Ec, wsao + (size_t)j*Ec*Ec, Ec, sabo + j*Ec, MTOT, Ec, Ec, yf, Ec, nullptr, 0, 0, 0);
      resid_ln_kernel<<<MTOT, 256, 0, stream>>>(xf, yf, salng + j*Ec, salnb + j*Ec, xbf);
    }
    // cross-attention: Q (padded 60->64 rows), K, V^T projections (WMMA)
    const unsigned short* Wc = wcaqkv + (size_t)l * 3 * Ec * Ec;
    launch_gemm(stream, xbf, Ec, Wc,                   Ec, cabqkv + l*3*Ec,        MTOT,  Ec, Ec, nullptr, 0, qpad, Ec,   1, 0);
    launch_gemm(stream, traj_bf, Ec, Wc + (size_t)Ec*Ec,   Ec, cabqkv + l*3*Ec + Ec,   MTRAJ, Ec, Ec, nullptr, 0, k_bf, Ec,   0, 0);
    launch_gemm(stream, traj_bf, Ec, Wc + (size_t)2*Ec*Ec, Ec, cabqkv + l*3*Ec + 2*Ec, MTRAJ, Ec, Ec, nullptr, 0, vt_bf, TLPc, 2, 0);
    attn_score_kernel<<<dim3(TLPc/16, QLPc/16, BGc*Hc), 32, 0, stream>>>(
        qpad, k_bf, tt, ut, tm, rpe + (size_t)l * PEc * Hc, scores);
    softmax_kernel<<<BGc * Hc * QLc, 256, 0, stream>>>(scores, p_bf);
    attn_pv_kernel<<<dim3(Dc/16, QLPc/16, BGc*Hc), 32, 0, stream>>>(p_bf, vt_bf, obf);
    launch_gemm(stream, obf, Ec, wcao + (size_t)l*Ec*Ec, Ec, cabo + l*Ec, MTOT, Ec, Ec, yf, Ec, nullptr, 0, 0, 0);
    resid_ln_kernel<<<MTOT, 256, 0, stream>>>(xf, yf, calng + l*Ec, calnb + l*Ec, xbf);
    // FFN (WMMA, fused exact GELU)
    launch_gemm(stream, xbf, Ec, wf1 + (size_t)l*DFFc*Ec, Ec, fb1 + l*DFFc, MTOT, DFFc, Ec, nullptr, 0, h1bf, DFFc, 0, 1);
    launch_gemm(stream, h1bf, DFFc, wf2 + (size_t)l*Ec*DFFc, DFFc, fb2 + l*Ec, MTOT, Ec, DFFc, yf, Ec, nullptr, 0, 0, 0);
    resid_ln_kernel<<<MTOT, 256, 0, stream>>>(xf, yf, flng + l*Ec, flnb + l*Ec, xbf);
  }
  head_kernel<<<MTOT, 64, 0, stream>>>(xf, Wout, out);
}
